// ChildSumTreeLSTM_81535659147895
// MI455X (gfx1250) — matrix operand order repaired and checked by reading, not compiled, not run on hardware
//
#include <hip/hip_runtime.h>

// ---------------- types ----------------
typedef __attribute__((ext_vector_type(16))) __bf16 v16bf;
typedef __attribute__((ext_vector_type(8)))  __bf16 v8bf;
typedef __attribute__((ext_vector_type(8)))  float  v8f;

#define MEM 256
#define INP 300
#define KA  576          // packed A row pitch: 300 (X) + 256 (child-sum h) + 20 pad
#define APITCH 72        // LDS row pitch (elems): 2 x 32-col K-tiles + 8 pad

static __device__ __forceinline__ __bf16 f2bf(float f) {
  unsigned u = __builtin_bit_cast(unsigned, f);
  u += 0x7fffu + ((u >> 16) & 1u);            // round-to-nearest-even
  unsigned short hs = (unsigned short)(u >> 16);
  return __builtin_bit_cast(__bf16, hs);
}
static __device__ __forceinline__ float sigm(float x) { return 1.0f / (1.0f + __expf(-x)); }

// ---------------- weight packing ----------------
// Bcat[1024 x 576] bf16, row n: n>>8 -> gate (0=i,1=o,2=u,3=f), n&255 -> m
// cols 0..299 = W{g}x[m], cols 300..555 = W{g}h[m] (zero for f gate), 556..575 = 0
__global__ void __launch_bounds__(256) pack_b(
    const float* __restrict__ Wix, const float* __restrict__ Wox,
    const float* __restrict__ Wux, const float* __restrict__ Wfx,
    const float* __restrict__ Wih, const float* __restrict__ Woh,
    const float* __restrict__ Wuh, __bf16* __restrict__ Bcat) {
  int idx = blockIdx.x * 256 + threadIdx.x;
  if (idx >= 1024 * KA) return;
  int n = idx / KA, k = idx % KA;
  int gate = n >> 8, m = n & 255;
  float v = 0.0f;
  if (k < INP) {
    const float* Wx = (gate == 0) ? Wix : (gate == 1) ? Wox : (gate == 2) ? Wux : Wfx;
    v = Wx[m * INP + k];
  } else if (k < INP + MEM) {
    int kh = k - INP;
    if (gate == 0) v = Wih[m * MEM + kh];
    else if (gate == 1) v = Woh[m * MEM + kh];
    else if (gate == 2) v = Wuh[m * MEM + kh];
  }
  Bcat[idx] = f2bf(v);
}

__global__ void __launch_bounds__(256) pack_wfh(const float* __restrict__ Wfh,
                                                __bf16* __restrict__ WfhB) {
  int idx = blockIdx.x * 256 + threadIdx.x;
  if (idx < MEM * MEM) WfhB[idx] = f2bf(Wfh[idx]);
}

// ---------------- fused A-operand pack for one level chunk ----------------
// Abf[r*576 + k]: k<300 -> X[(c0+r)*300+k]; k<556 -> sum of children h; else 0.
// Leaves get zero child-sum automatically (4p+1 >= N).
__global__ void __launch_bounds__(256) pack_a(
    const float* __restrict__ X, const float* __restrict__ h,
    __bf16* __restrict__ Abf, int c0, int rows, int N) {
  int idx = blockIdx.x * 256 + threadIdx.x;
  if (idx >= rows * KA) return;
  int r = idx / KA, k = idx % KA;
  int p = c0 + r;
  float v = 0.0f;
  if (k < INP) {
    v = X[(size_t)p * INP + k];
  } else if (k < INP + MEM) {
    int m = k - INP;
    int ch0 = 4 * p + 1;
#pragma unroll
    for (int j = 0; j < 4; ++j) {
      int ch = ch0 + j;
      if (ch < N) v += h[(size_t)ch * MEM + m];
    }
  }
  Abf[idx] = f2bf(v);
}

// ---------------- WMMA GEMM:  C[M x (gridDim.y*64)] = A[M x K] * B^T ----------------
// A bf16 row-major [M x lda]; B bf16 row-major [Ncols x ldb] (row = output col).
// kTiles MUST be even. A tiles are DMA'd into double-buffered LDS via
// GLOBAL_LOAD_ASYNC_TO_LDS_B128 (ASYNCcnt), overlapping the WMMAs of the
// previous tile. Rows >= M are clamped to row M-1 (an A row only feeds the
// same C row, and those C rows are masked in the epilogue).
__global__ void __launch_bounds__(256) wmma_gemm(
    const __bf16* __restrict__ A, int lda,
    const __bf16* __restrict__ B, int ldb,
    float* __restrict__ C, int ldc,
    int M, int kTiles) {
  const int lane = threadIdx.x & 31;
  const int w    = threadIdx.x >> 5;
  const int wm   = w >> 1;          // 0..3  (M sub-block of 32)
  const int wn   = w & 1;           // 0..1  (N sub-block of 32)
  const int mb   = blockIdx.x * 128;
  const int nb   = blockIdx.y * 64;

  // double-buffered: 128 rows x 64 cols (two 32-col K-tiles), pitch 72 elems
  __shared__ __bf16 As[2][128 * APITCH];

  v8f acc00 = {}, acc01 = {}, acc10 = {}, acc11 = {};

  const int koB   = (lane >> 4) * 16;   // B: lanes 0-15 -> K 0..15, lanes 16-31 -> K 16..31
  const int koA   = (lane >> 4) * 16;   // A: lanes 0-15 -> K 0..15, lanes 16-31 -> K 16..31
  const int nrow0 = nb + wn * 32 + (lane & 15);
  const int ar0   = wm * 32 + (lane & 15);

  // staging role: thread t covers row r = t>>1, 16-col half of each 32-col tile
  const int sr    = threadIdx.x >> 1;
  const int shalf = (threadIdx.x & 1) * 16;
  int gr = mb + sr;
  if (gr >= M) gr = M - 1;                       // clamp (see note above)
  const __bf16* aRow = A + (size_t)gr * lda + shalf;
  const unsigned lds0 = (unsigned)(size_t)(&As[0][sr * APITCH + shalf]);
  const unsigned lds1 = (unsigned)(size_t)(&As[1][sr * APITCH + shalf]);

  // async DMA of one 128x64 A block (4 x B128 per thread; same IOFFSET applies
  // to both LDS and global sides, and the layouts advance identically)
#define ASYNC_A_TILE(LDSA, GA)                                              \
  asm volatile("global_load_async_to_lds_b128 %0, %1, off\n\t"              \
               "global_load_async_to_lds_b128 %0, %1, off offset:16\n\t"    \
               "global_load_async_to_lds_b128 %0, %1, off offset:64\n\t"    \
               "global_load_async_to_lds_b128 %0, %1, off offset:80"        \
               :: "v"(LDSA), "v"((unsigned long long)(size_t)(GA))          \
               : "memory")

  ASYNC_A_TILE(lds0, aRow);                      // prologue: tiles {0,1} -> buf 0

  for (int kt = 0; kt < kTiles; kt += 2) {
    const int kb  = kt * 32;
    const int cur = (kt >> 1) & 1;

    asm volatile("s_wait_asynccnt 0" ::: "memory");   // our DMA (buf cur) landed
    __syncthreads();                                  // everyone's DMA landed; prev reads done

    if (kt + 2 < kTiles) {                            // DMA next block into other buffer
      const unsigned nextLds = cur ? lds0 : lds1;     // address select, no branch
      ASYNC_A_TILE(nextLds, aRow + kb + 64);
    }

    // B operands straight from global (32B aligned, L2-resident weights)
    const __bf16* bRow0 = B + (size_t)nrow0 * ldb + kb + koB;
    const __bf16* bRow1 = B + (size_t)(nrow0 + 16) * ldb + kb + koB;
    v16bf b0t0 = *(const v16bf*)(bRow0);
    v16bf b0t1 = *(const v16bf*)(bRow0 + 32);
    v16bf b1t0 = *(const v16bf*)(bRow1);
    v16bf b1t1 = *(const v16bf*)(bRow1 + 32);
    if (kt + 2 < kTiles) __builtin_prefetch((const void*)(bRow0 + 64), 0, 0);

    const __bf16* lbase = &As[cur][0];
    // A operands from LDS in WMMA lane layout (tile kt)
    v8bf x0l = *(const v8bf*)(lbase + ar0 * APITCH + koA);
    v8bf x0h = *(const v8bf*)(lbase + ar0 * APITCH + 8 + koA);
    v8bf x1l = *(const v8bf*)(lbase + (ar0 + 16) * APITCH + koA);
    v8bf x1h = *(const v8bf*)(lbase + (ar0 + 16) * APITCH + 8 + koA);
    v16bf a0 = __builtin_shufflevector(x0l, x0h, 0,1,2,3,4,5,6,7,8,9,10,11,12,13,14,15);
    v16bf a1 = __builtin_shufflevector(x1l, x1h, 0,1,2,3,4,5,6,7,8,9,10,11,12,13,14,15);

    acc00 = __builtin_amdgcn_wmma_f32_16x16x32_bf16(false, a0, false, b0t0, (short)0, acc00, false, false);
    acc01 = __builtin_amdgcn_wmma_f32_16x16x32_bf16(false, a0, false, b1t0, (short)0, acc01, false, false);
    acc10 = __builtin_amdgcn_wmma_f32_16x16x32_bf16(false, a1, false, b0t0, (short)0, acc10, false, false);
    acc11 = __builtin_amdgcn_wmma_f32_16x16x32_bf16(false, a1, false, b1t0, (short)0, acc11, false, false);

    // A operands (tile kt+1) at LDS col offset 32
    v8bf y0l = *(const v8bf*)(lbase + ar0 * APITCH + 32 + koA);
    v8bf y0h = *(const v8bf*)(lbase + ar0 * APITCH + 40 + koA);
    v8bf y1l = *(const v8bf*)(lbase + (ar0 + 16) * APITCH + 32 + koA);
    v8bf y1h = *(const v8bf*)(lbase + (ar0 + 16) * APITCH + 40 + koA);
    v16bf c0v = __builtin_shufflevector(y0l, y0h, 0,1,2,3,4,5,6,7,8,9,10,11,12,13,14,15);
    v16bf c1v = __builtin_shufflevector(y1l, y1h, 0,1,2,3,4,5,6,7,8,9,10,11,12,13,14,15);

    acc00 = __builtin_amdgcn_wmma_f32_16x16x32_bf16(false, c0v, false, b0t1, (short)0, acc00, false, false);
    acc01 = __builtin_amdgcn_wmma_f32_16x16x32_bf16(false, c0v, false, b1t1, (short)0, acc01, false, false);
    acc10 = __builtin_amdgcn_wmma_f32_16x16x32_bf16(false, c1v, false, b0t1, (short)0, acc10, false, false);
    acc11 = __builtin_amdgcn_wmma_f32_16x16x32_bf16(false, c1v, false, b1t1, (short)0, acc11, false, false);
  }
#undef ASYNC_A_TILE

  // epilogue: C 16x16 f32 layout -> VGPR v: row = v + 8*(lane>>4), col = lane&15
  const int cr = (lane >> 4) * 8;
  const int cc = lane & 15;
#pragma unroll
  for (int v = 0; v < 8; ++v) {
    int r0 = mb + wm * 32 + cr + v;
    int c0 = nb + wn * 32 + cc;
    if (r0 < M) {
      C[(size_t)r0 * ldc + c0]      = acc00[v];
      C[(size_t)r0 * ldc + c0 + 16] = acc01[v];
    }
    int r1 = r0 + 16;
    if (r1 < M) {
      C[(size_t)r1 * ldc + c0]      = acc10[v];
      C[(size_t)r1 * ldc + c0 + 16] = acc11[v];
    }
  }
}

// ---------------- gate combine ----------------
__global__ void __launch_bounds__(256) combine(
    const float* __restrict__ G, const float* __restrict__ F,
    const float* __restrict__ bix, const float* __restrict__ box,
    const float* __restrict__ bux, const float* __restrict__ bfx,
    const float* __restrict__ fbv,
    float* __restrict__ h, __bf16* __restrict__ hbf, float* __restrict__ cbuf,
    int s, int rows, int cs, int N) {
  int l = blockIdx.x, m = threadIdx.x;
  if (l >= rows) return;
  int p = s + l;
  const float* g = G + (size_t)l * 1024;
  float i = sigm(g[m] + bix[m]);
  float o = sigm(g[256 + m] + box[m]);
  float u = tanhf(g[512 + m] + bux[m]);
  float fc = 0.0f;
  int ch0 = 4 * p + 1;
  if (ch0 < N) {
    float fx = g[768 + m] + bfx[m] + fbv[m];
#pragma unroll
    for (int k = 0; k < 4; ++k) {
      int ch = ch0 + k;
      if (ch < N) {
        float f = sigm(F[(size_t)(ch - cs) * MEM + m] + fx);
        fc += f * cbuf[(size_t)ch * MEM + m];
      }
    }
  }
  float cn = i * u + fc;
  float hn = o * tanhf(cn);
  cbuf[(size_t)p * MEM + m] = cn;
  h[(size_t)p * MEM + m]    = hn;
  hbf[(size_t)p * MEM + m]  = f2bf(hn);   // bf16 copy feeds the next level's Wfh GEMM
}

__global__ void __launch_bounds__(256) root_out(const float* __restrict__ h,
                                                const float* __restrict__ cbuf,
                                                float* __restrict__ out) {
  int t = blockIdx.x * 256 + threadIdx.x;
  if (t < 256) out[t] = h[t];
  else if (t < 512) out[t] = cbuf[t - 256];
}

// ---------------- host orchestration ----------------
extern "C" void kernel_launch(void* const* d_in, const int* in_sizes, int n_in,
                              void* d_out, int out_size, void* d_ws, size_t ws_size,
                              hipStream_t stream) {
  const float* X   = (const float*)d_in[0];
  const float* Wix = (const float*)d_in[2];
  const float* bix = (const float*)d_in[3];
  const float* Wfx = (const float*)d_in[4];
  const float* bfx = (const float*)d_in[5];
  const float* Wox = (const float*)d_in[6];
  const float* box = (const float*)d_in[7];
  const float* Wux = (const float*)d_in[8];
  const float* bux = (const float*)d_in[9];
  const float* Wih = (const float*)d_in[10];
  const float* Woh = (const float*)d_in[11];
  const float* Wuh = (const float*)d_in[12];
  const float* Wfh = (const float*)d_in[13];
  const float* fbv = (const float*)d_in[14];
  const int N = in_sizes[1];     // number of nodes (parent array length)

  float* out = (float*)d_out;
  float* h   = out + 512;        // all-node hidden states live directly in d_out

  // ---- carve workspace ----
  char* base = (char*)d_ws;
  size_t off = 0;
  auto carve = [&](size_t bytes) -> void* {
    off = (off + 255) & ~(size_t)255;
    void* p = base + off;
    off += bytes;
    return p;
  };
  __bf16* Bcat = (__bf16*)carve((size_t)1024 * KA * 2);
  __bf16* WfhB = (__bf16*)carve((size_t)MEM * MEM * 2);
  float*  cbuf = (float*)carve((size_t)N * MEM * 4);
  __bf16* hbf  = (__bf16*)carve((size_t)N * MEM * 2);
  size_t fixedEnd = off;
  // per chunk-row: Abf (576 bf16) + G (1024 f32) + F (4*256 f32)
  const size_t perRow = (size_t)KA * 2 + 1024 * 4 + 4 * MEM * 4;
  long long chunk = 8192;
  if (ws_size > fixedEnd + 65536) {
    long long maxc = (long long)((ws_size - fixedEnd - 4096) / perRow);
    if (maxc < chunk) chunk = maxc;
  } else {
    chunk = 128;
  }
  if (chunk < 1) chunk = 1;
  __bf16* Abf = (__bf16*)carve((size_t)chunk * KA * 2);
  float*  G   = (float*)carve((size_t)chunk * 1024 * 4);
  float*  F   = (float*)carve((size_t)chunk * 4 * MEM * 4);

  // ---- pack weights to bf16 (every call; deterministic) ----
  pack_b<<<(1024 * KA + 255) / 256, 256, 0, stream>>>(Wix, Wox, Wux, Wfx, Wih, Woh, Wuh, Bcat);
  pack_wfh<<<(MEM * MEM + 255) / 256, 256, 0, stream>>>(Wfh, WfhB);

  // ---- level boundaries: depth-d nodes are [(4^d-1)/3, (4^{d+1}-1)/3) ----
  long long starts[34];
  int nd = 0;
  starts[0] = 0;
  while (nd < 32 && starts[nd] < (long long)N) { starts[nd + 1] = 4 * starts[nd] + 1; ++nd; }

  for (int d = nd - 1; d >= 0; --d) {   // deepest level first
    long long s = starts[d];
    long long e = (starts[d + 1] < (long long)N) ? starts[d + 1] : (long long)N;
    for (long long c0 = s; c0 < e; c0 += chunk) {
      long long c1 = (c0 + chunk < e) ? c0 + chunk : e;
      int rows = (int)(c1 - c0);
      long long chs = 4 * c0 + 1;
      long long che = (4 * c1 + 1 < (long long)N) ? 4 * c1 + 1 : (long long)N;
      bool kids = che > chs;

      // fused A pack: [X | child-sum h] in bf16 (child-sum is 0 for leaves)
      pack_a<<<(rows * KA + 255) / 256, 256, 0, stream>>>(X, h, Abf, (int)c0, rows, N);

      if (kids) {
        int crows = (int)(che - chs);
        // F = h[children] * Wfh^T   (K=256 -> 8 k-tiles, N=256 -> grid.y=4)
        wmma_gemm<<<dim3((crows + 127) / 128, 4), 256, 0, stream>>>(
            hbf + (size_t)chs * MEM, MEM, WfhB, MEM, F, MEM, crows, 8);
        // G = [X | hs] * Bcat^T     (K=576 -> 18 k-tiles, N=1024 -> grid.y=16)
        wmma_gemm<<<dim3((rows + 127) / 128, 16), 256, 0, stream>>>(
            Abf, KA, Bcat, KA, G, 1024, rows, 18);
      } else {
        // leaf chunk: child-sum cols are zero -> only K=320 (10 k-tiles) needed
        wmma_gemm<<<dim3((rows + 127) / 128, 16), 256, 0, stream>>>(
            Abf, KA, Bcat, KA, G, 1024, rows, 10);
      }
      combine<<<rows, 256, 0, stream>>>(G, F, bix, box, bux, bfx, fbv,
                                        h, hbf, cbuf, (int)c0, rows, (int)chs, N);
    }
  }
  root_out<<<2, 256, 0, stream>>>(h, cbuf, out);
}
